// MoE_75239237091571
// MI455X (gfx1250) — matrix operand order repaired and checked by reading, not compiled, not run on hardware
//
#include <hip/hip_runtime.h>
#include <hip/hip_bf16.h>

// ---------------------------------------------------------------------------
// MoE NeRF forward for gfx1250 (MI455X). All GEMMs on v_wmma_f32_16x16x32_f16.
// ---------------------------------------------------------------------------

typedef __attribute__((ext_vector_type(16))) _Float16 v16h;
typedef __attribute__((ext_vector_type(8)))  float    v8f;

#define WMMA_F16(a, b, c) \
  __builtin_amdgcn_wmma_f32_16x16x32_f16(false, (a), false, (b), (short)0, (c), false, false)

// A-fragment (16x32 f16, M x K) gather from row-major LDS.
// Contiguous k order used here (same as B packing): lane = hi*16 + m ; half-slot e -> K = 16*hi + e
__device__ __forceinline__ v16h frag_a_lds(const _Float16* base, int ld, int lane) {
  const int r = lane & 15, hi = lane >> 4;
  const _Float16* row = base + r * ld;
  v16h a;
#pragma unroll
  for (int v = 0; v < 8; ++v) {
    int e = v * 2;
    int k = hi * 16 + e;   // even k -> pairs merge to b32 ds loads
    a[e]     = row[k];
    a[e + 1] = row[k + 1];
  }
  return a;
}

// ---------------------------------------------------------------------------
// Weight conversion: f32 [M][Ksrc][256] -> f16, K zero-padded to Kdst (mult of
// 32), stored in B-fragment order: flat half index
//   (((m*KT + kt)*16 + nt)*32 + lane)*16 + e
// with K = kt*32 + (lane>>4)*16 + e, N = nt*16 + (lane&15).
// Consumption is then a single aligned v16h (32B) load per lane per fragment.
// ---------------------------------------------------------------------------
__global__ void moe_convert_kernel(const float* __restrict__ src,
                                   _Float16* __restrict__ dst,
                                   int M, int Ksrc, int Kdst) {
  const int KT = Kdst >> 5;
  const long total = (long)M * Kdst * 256;
  for (long i = (long)blockIdx.x * blockDim.x + threadIdx.x; i < total;
       i += (long)gridDim.x * blockDim.x) {
    int e2   = (int)(i & 15);
    int lane = (int)((i >> 4) & 31);
    long t   = i >> 9;
    int nt   = (int)(t & 15);
    long t2  = t >> 4;
    int kt   = (int)(t2 % KT);
    int m    = (int)(t2 / KT);
    int k  = kt * 32 + (lane >> 4) * 16 + e2;
    int nn = nt * 16 + (lane & 15);
    float v = (k < Ksrc) ? src[((long)m * Ksrc + k) * 256 + nn] : 0.f;
    dst[i] = (_Float16)v;
  }
}

// ---------------------------------------------------------------------------
// Gate MLP: [B,319] -> relu -> [256] -> [256] -> LN -> [8] -> softmax -> top1
// One block = 16 points, 512 threads = 16 waves, wave w owns columns 16w..16w+15.
// ---------------------------------------------------------------------------
__global__ __launch_bounds__(512) void moe_gate_kernel(
    const float* __restrict__ x, const float* __restrict__ lat,
    const _Float16* __restrict__ gw1h, const float* __restrict__ gb1,
    const _Float16* __restrict__ gw2h, const float* __restrict__ gb2,
    const float* __restrict__ lng, const float* __restrict__ lnb,
    const float* __restrict__ gw3, const float* __restrict__ gb3,
    int* __restrict__ eid, float* __restrict__ gval) {
  __shared__ _Float16 Ain[16 * 320];
  __shared__ _Float16 H1[16 * 256];
  __shared__ float    H2[16 * 256];
  __shared__ float    LG[16 * 8];
  const int tid  = threadIdx.x;
  const int lane = tid & 31;
  const int w    = tid >> 5;
  const int p0   = blockIdx.x * 16;

  // stage A = [x || latent], zero-pad K col 319   (5120 elems -> exactly 10 iters)
  for (int i = tid; i < 16 * 320; i += 512) {
    int r = i / 320, c = i - r * 320;
    float v = 0.f;
    if (c < 63)       v = x[(p0 + r) * 63 + c];
    else if (c < 319) v = lat[(p0 + r) * 256 + (c - 63)];
    Ain[i] = (_Float16)v;
  }
  __syncthreads();

  const int n  = w * 16 + (lane & 15);
  const int hi = lane >> 4;

  { // h1 = relu(A @ gw1 + gb1)   K = 320
    v8f acc = {};
    const v16h* wb = (const v16h*)gw1h;
#pragma unroll
    for (int kt = 0; kt < 10; ++kt) {
      v16h a = frag_a_lds(Ain + kt * 32, 320, lane);
      v16h b = wb[(kt * 16 + w) * 32 + lane];
      acc = WMMA_F16(a, b, acc);
    }
    float bv = gb1[n];
#pragma unroll
    for (int vi = 0; vi < 8; ++vi) {
      float val = acc[vi] + bv;
      H1[(vi + 8 * hi) * 256 + n] = (_Float16)(val > 0.f ? val : 0.f);
    }
  }
  __syncthreads();

  { // h2 = h1 @ gw2 + gb2   K = 256 (f32 for LayerNorm)
    v8f acc = {};
    const v16h* wb = (const v16h*)gw2h;
#pragma unroll
    for (int kt = 0; kt < 8; ++kt) {
      v16h a = frag_a_lds(H1 + kt * 32, 256, lane);
      v16h b = wb[(kt * 16 + w) * 32 + lane];
      acc = WMMA_F16(a, b, acc);
    }
    float bv = gb2[n];
#pragma unroll
    for (int vi = 0; vi < 8; ++vi)
      H2[(vi + 8 * hi) * 256 + n] = acc[vi] + bv;
  }
  __syncthreads();

  // LayerNorm, one thread per row
  if (tid < 16) {
    float mu = 0.f;
    for (int c = 0; c < 256; ++c) mu += H2[tid * 256 + c];
    mu *= (1.f / 256.f);
    float var = 0.f;
    for (int c = 0; c < 256; ++c) { float d = H2[tid * 256 + c] - mu; var += d * d; }
    var *= (1.f / 256.f);
    float rs = rsqrtf(var + 1e-5f);
    for (int c = 0; c < 256; ++c)
      H2[tid * 256 + c] = (H2[tid * 256 + c] - mu) * rs * lng[c] + lnb[c];
  }
  __syncthreads();

  // logits = hn @ gw3 + gb3  (256 -> 8): one thread per (row, expert)
  if (tid < 128) {
    int r = tid >> 3, e = tid & 7;
    float s = gb3[e];
    for (int c = 0; c < 256; ++c) s += H2[r * 256 + c] * gw3[c * 8 + e];
    LG[r * 8 + e] = s;
  }
  __syncthreads();

  // softmax + argmax (first max wins, like jnp.argmax)
  if (tid < 16) {
    float mx = LG[tid * 8]; int am = 0;
    for (int e = 1; e < 8; ++e) { float v = LG[tid * 8 + e]; if (v > mx) { mx = v; am = e; } }
    float se = 0.f;
    for (int e = 0; e < 8; ++e) se += expf(LG[tid * 8 + e] - mx);
    eid[p0 + tid]  = am;
    gval[p0 + tid] = 1.f / se;   // exp(mx - mx) / sum
  }
}

// ---------------------------------------------------------------------------
// Routing: histogram -> scan -> scatter.  ctrl layout (ints):
//   [0..7]   per-expert counts
//   [8..16]  row offsets (exclusive scan, 9 entries)
//   [17..25] tile offsets (9 entries)
//   [26..33] scatter cursors
// ---------------------------------------------------------------------------
__global__ void moe_zero_kernel(int* __restrict__ ctrl) {
  if (threadIdx.x < 34) ctrl[threadIdx.x] = 0;
}

__global__ void moe_hist_kernel(const int* __restrict__ eid, int* __restrict__ ctrl) {
  int p = blockIdx.x * blockDim.x + threadIdx.x;
  if (p < 16384) atomicAdd(&ctrl[eid[p]], 1);
}

__global__ void moe_scan_kernel(int* __restrict__ ctrl) {
  if (threadIdx.x == 0 && blockIdx.x == 0) {
    int off = 0, toff = 0;
    ctrl[8] = 0; ctrl[17] = 0;
    for (int e = 0; e < 8; ++e) {
      ctrl[26 + e] = off;
      off  += ctrl[e];                 ctrl[8 + e + 1]  = off;
      toff += (ctrl[e] + 15) >> 4;     ctrl[17 + e + 1] = toff;
    }
  }
}

__global__ void moe_scatter_kernel(const int* __restrict__ eid,
                                   int* __restrict__ ctrl, int* __restrict__ perm) {
  int p = blockIdx.x * blockDim.x + threadIdx.x;
  if (p < 16384) {
    int pos = atomicAdd(&ctrl[26 + eid[p]], 1);
    perm[pos] = p;
  }
}

// ---------------------------------------------------------------------------
// Expert NeRF MLP, one block per 16-point tile of one expert.
// 512 threads = 16 waves; wave w owns hidden columns 16w..16w+15.
// h0(95,pad96) -> L0 -> L1..L4 -> L5(skip, K=352=act||h0) -> L6 -> out(4)
// ---------------------------------------------------------------------------
__global__ __launch_bounds__(512) void moe_expert_kernel(
    const float* __restrict__ x, const float* __restrict__ lat,
    const int* __restrict__ perm, const int* __restrict__ ctrl,
    const float* __restrict__ gval,
    const _Float16* __restrict__ ew0h, const _Float16* __restrict__ ewmh,
    const _Float16* __restrict__ ew5h,
    const float* __restrict__ eb0, const float* __restrict__ eb1,
    const float* __restrict__ eb2, const float* __restrict__ eb3,
    const float* __restrict__ eb4, const float* __restrict__ eb5,
    const float* __restrict__ eb6,
    const float* __restrict__ ewo, const float* __restrict__ ebo,
    float* __restrict__ out) {
  __shared__ _Float16 H0[16 * 96];
  __shared__ _Float16 Abuf[2][16 * 256];   // ping-pong activations (no ptr-array init)
  __shared__ int rowsP[16];
  __shared__ int sInfo[3];   // expert, row base, nvalid
  const int tid  = threadIdx.x;
  const int lane = tid & 31;
  const int w    = tid >> 5;

  if (tid == 0) {
    int b = blockIdx.x;
    int e = -1, base = 0, nv = 0;
    if (b < ctrl[17 + 8]) {
      for (int i = 0; i < 8; ++i)
        if (b >= ctrl[17 + i] && b < ctrl[17 + i + 1]) { e = i; break; }
      int lt = b - ctrl[17 + e];
      base = ctrl[8 + e] + 16 * lt;
      nv = ctrl[e] - 16 * lt; if (nv > 16) nv = 16;
    }
    sInfo[0] = e; sInfo[1] = base; sInfo[2] = nv;
  }
  __syncthreads();
  const int e = sInfo[0];
  if (e < 0) return;                      // uniform exit for padding blocks
  const int base = sInfo[1], nv = sInfo[2];
  if (tid < 16) rowsP[tid] = perm[base + (tid < nv ? tid : nv - 1)];
  __syncthreads();

  // stage h0 = [x(63) || latent chunk e (32)], col 95 zero-pad (1536 -> 3 iters)
  for (int i = tid; i < 16 * 96; i += 512) {
    int r = i / 96, c = i - r * 96;
    int p = rowsP[r];
    float v = 0.f;
    if (c < 63)      v = x[p * 63 + c];
    else if (c < 95) v = lat[p * 256 + e * 32 + (c - 63)];
    H0[i] = (_Float16)v;
  }
  __syncthreads();

  const int n  = w * 16 + (lane & 15);
  const int hi = lane >> 4;
  int cur = 0;

  { // layer 0: K = 96
    v8f acc = {};
    const v16h* wb = (const v16h*)ew0h + (long)e * (3 * 16 * 32);
#pragma unroll
    for (int kt = 0; kt < 3; ++kt) {
      v16h a = frag_a_lds(H0 + kt * 32, 96, lane);
      v16h b = wb[(kt * 16 + w) * 32 + lane];
      acc = WMMA_F16(a, b, acc);
    }
    float bv = eb0[e * 256 + n];
#pragma unroll
    for (int vi = 0; vi < 8; ++vi) {
      float val = acc[vi] + bv;
      Abuf[0][(vi + 8 * hi) * 256 + n] = (_Float16)(val > 0.f ? val : 0.f);
    }
  }
  __syncthreads();

  // layers 1..4 (ewmh slots 0..3), ping-pong LDS activations
  for (int l = 0; l < 4; ++l) {
    const float* bias = (l == 0) ? eb1 : (l == 1) ? eb2 : (l == 2) ? eb3 : eb4;
    v8f acc = {};
    const v16h* wb = (const v16h*)ewmh + ((long)l * 8 + e) * (8 * 16 * 32);
#pragma unroll
    for (int kt = 0; kt < 8; ++kt) {
      v16h a = frag_a_lds(&Abuf[cur][kt * 32], 256, lane);
      v16h b = wb[(kt * 16 + w) * 32 + lane];
      acc = WMMA_F16(a, b, acc);
    }
    float bv = bias[e * 256 + n];
    _Float16* dst = Abuf[cur ^ 1];
#pragma unroll
    for (int vi = 0; vi < 8; ++vi) {
      float val = acc[vi] + bv;
      dst[(vi + 8 * hi) * 256 + n] = (_Float16)(val > 0.f ? val : 0.f);
    }
    __syncthreads();
    cur ^= 1;
  }

  { // layer 5 (skip): K = 352 = act(256) || h0(95, pad)
    v8f acc = {};
    const v16h* wb = (const v16h*)ew5h + (long)e * (11 * 16 * 32);
#pragma unroll
    for (int kt = 0; kt < 8; ++kt) {
      v16h a = frag_a_lds(&Abuf[cur][kt * 32], 256, lane);
      v16h b = wb[(kt * 16 + w) * 32 + lane];
      acc = WMMA_F16(a, b, acc);
    }
#pragma unroll
    for (int kt = 0; kt < 3; ++kt) {
      v16h a = frag_a_lds(H0 + kt * 32, 96, lane);
      v16h b = wb[((8 + kt) * 16 + w) * 32 + lane];
      acc = WMMA_F16(a, b, acc);
    }
    float bv = eb5[e * 256 + n];
    _Float16* dst = Abuf[cur ^ 1];
#pragma unroll
    for (int vi = 0; vi < 8; ++vi) {
      float val = acc[vi] + bv;
      dst[(vi + 8 * hi) * 256 + n] = (_Float16)(val > 0.f ? val : 0.f);
    }
    __syncthreads();
    cur ^= 1;
  }

  { // layer 6 (ewmh slot 4)
    v8f acc = {};
    const v16h* wb = (const v16h*)ewmh + ((long)4 * 8 + e) * (8 * 16 * 32);
#pragma unroll
    for (int kt = 0; kt < 8; ++kt) {
      v16h a = frag_a_lds(&Abuf[cur][kt * 32], 256, lane);
      v16h b = wb[(kt * 16 + w) * 32 + lane];
      acc = WMMA_F16(a, b, acc);
    }
    float bv = eb6[e * 256 + n];
    _Float16* dst = Abuf[cur ^ 1];
#pragma unroll
    for (int vi = 0; vi < 8; ++vi) {
      float val = acc[vi] + bv;
      dst[(vi + 8 * hi) * 256 + n] = (_Float16)(val > 0.f ? val : 0.f);
    }
    __syncthreads();
    cur ^= 1;
  }

  // output head 256 -> 4, fused log(gate * exp(y)) combine
  if (tid < 64) {
    int r = tid >> 2, o = tid & 3;
    if (r < nv) {
      const _Float16* act = Abuf[cur];
      float s = ebo[e * 4 + o];
      for (int k = 0; k < 256; ++k)
        s += (float)act[r * 256 + k] * ewo[((long)e * 256 + k) * 4 + o];
      int p = rowsP[r];
      float c = gval[p] * expf(s);
      if (c == 0.f) c = 2.220446049250313e-16f;   // np eps clamp from reference
      out[p * 4 + o] = logf(c);
    }
  }
}

// ---------------------------------------------------------------------------
// Host launcher. Workspace layout (bytes):
//   gw1h  [320*256]h      @ 0
//   gw2h  [256*256]h      @ 163840
//   ew0h  [8*96*256]h     @ 294912
//   ewmh  [5*8*256*256]h  @ 688128     (slots: ew1,ew2,ew3,ew4,ew6)
//   ew5h  [8*352*256]h    @ 5931008
//   eid   [16384]i        @ 7372800
//   gval  [16384]f        @ 7438336
//   perm  [16384]i        @ 7503872
//   ctrl  [34]i           @ 7569408
// ---------------------------------------------------------------------------
extern "C" void kernel_launch(void* const* d_in, const int* in_sizes, int n_in,
                              void* d_out, int out_size, void* d_ws, size_t ws_size,
                              hipStream_t stream) {
  (void)in_sizes; (void)n_in; (void)out_size; (void)ws_size;
  const float* x   = (const float*)d_in[0];
  const float* lat = (const float*)d_in[1];
  const float* gw1 = (const float*)d_in[2];
  const float* gb1 = (const float*)d_in[3];
  const float* gw2 = (const float*)d_in[4];
  const float* gb2 = (const float*)d_in[5];
  const float* lng = (const float*)d_in[6];
  const float* lnb = (const float*)d_in[7];
  const float* gw3 = (const float*)d_in[8];
  const float* gb3 = (const float*)d_in[9];
  const float* ew0 = (const float*)d_in[10];
  const float* eb0 = (const float*)d_in[11];
  const float* ew1 = (const float*)d_in[12];
  const float* eb1 = (const float*)d_in[13];
  const float* ew2 = (const float*)d_in[14];
  const float* eb2 = (const float*)d_in[15];
  const float* ew3 = (const float*)d_in[16];
  const float* eb3 = (const float*)d_in[17];
  const float* ew4 = (const float*)d_in[18];
  const float* eb4 = (const float*)d_in[19];
  const float* ew5 = (const float*)d_in[20];
  const float* eb5 = (const float*)d_in[21];
  const float* ew6 = (const float*)d_in[22];
  const float* eb6 = (const float*)d_in[23];
  const float* ewo = (const float*)d_in[24];
  const float* ebo = (const float*)d_in[25];
  float* out = (float*)d_out;

  char* ws = (char*)d_ws;
  _Float16* gw1h = (_Float16*)(ws + 0);
  _Float16* gw2h = (_Float16*)(ws + 163840);
  _Float16* ew0h = (_Float16*)(ws + 294912);
  _Float16* ewmh = (_Float16*)(ws + 688128);
  _Float16* ew5h = (_Float16*)(ws + 5931008);
  int*   eid  = (int*)(ws + 7372800);
  float* gv   = (float*)(ws + 7438336);
  int*   perm = (int*)(ws + 7503872);
  int*   ctrl = (int*)(ws + 7569408);

  auto cvt = [&](const float* src, _Float16* dst, int M, int Ksrc, int Kdst) {
    long total = (long)M * Kdst * 256;
    int blocks = (int)((total + 255) / 256);
    moe_convert_kernel<<<blocks, 256, 0, stream>>>(src, dst, M, Ksrc, Kdst);
  };

  // weight conversion to f16, B-fragment-swizzled, K zero-padded
  cvt(gw1, gw1h, 1, 319, 320);
  cvt(gw2, gw2h, 1, 256, 256);
  cvt(ew0, ew0h, 8,  95,  96);
  cvt(ew1, ewmh + 0L * 8 * 256 * 256, 8, 256, 256);
  cvt(ew2, ewmh + 1L * 8 * 256 * 256, 8, 256, 256);
  cvt(ew3, ewmh + 2L * 8 * 256 * 256, 8, 256, 256);
  cvt(ew4, ewmh + 3L * 8 * 256 * 256, 8, 256, 256);
  cvt(ew6, ewmh + 4L * 8 * 256 * 256, 8, 256, 256);
  cvt(ew5, ew5h, 8, 351, 352);

  // gate + top-1 routing
  moe_gate_kernel<<<1024, 512, 0, stream>>>(x, lat, gw1h, gb1, gw2h, gb2,
                                            lng, lnb, gw3, gb3, eid, gv);
  moe_zero_kernel<<<1, 64, 0, stream>>>(ctrl);
  moe_hist_kernel<<<64, 256, 0, stream>>>(eid, ctrl);
  moe_scan_kernel<<<1, 32, 0, stream>>>(ctrl);
  moe_scatter_kernel<<<64, 256, 0, stream>>>(eid, ctrl, perm);

  // routed expert MLP: max tiles = B/16 + E = 1032 (extras exit early)
  moe_expert_kernel<<<1032, 512, 0, stream>>>(x, lat, perm, ctrl, gv,
                                              ew0h, ewmh, ew5h,
                                              eb0, eb1, eb2, eb3, eb4, eb5, eb6,
                                              ewo, ebo, out);
}
